// SparseAttentionLayer_62139586839034
// MI455X (gfx1250) — matrix operand order repaired and checked by reading, compiled, not run on hardware
//
#include <hip/hip_runtime.h>
#include <hip/hip_bf16.h>
#include <stdint.h>

#define DEVI __device__ __forceinline__

typedef __bf16 bf16;
typedef __attribute__((ext_vector_type(16))) __bf16 v16bf;
typedef __attribute__((ext_vector_type(8)))  __bf16 v8bf;
typedef __attribute__((ext_vector_type(8)))  float  v8f;
typedef __attribute__((ext_vector_type(4)))  float  v4f;
typedef __attribute__((ext_vector_type(4)))  unsigned int v4u;
typedef __attribute__((ext_vector_type(8)))  int    v8i;
typedef __attribute__((ext_vector_type(4)))  int    v4i;

constexpr int   B_  = 8, L_ = 4096, NF = 256;
constexpr int   M_  = B_ * L_;                 // 32768 rows total
constexpr float NEGV = -1e30f;
constexpr float EPS  = 1e-4f;

// ---- workspace layout (bytes) ----
constexpr size_t OFF_STATS = 0;                                  // sum[256], sumsq[256], cnt @ idx 512
constexpr size_t OFF_WQT   = 4096;                               // 4 transposed bf16 weights
constexpr size_t OFF_QB    = OFF_WQT + 4 * (size_t)NF * NF * 2;  // q bf16 row-major [M][NF]
constexpr size_t OFF_KB    = OFF_QB  + (size_t)M_ * NF * 2;      // k bf16 row-major [M][NF]
constexpr size_t OFF_VT    = OFF_KB  + (size_t)M_ * NF * 2;      // v bf16 transposed [B][NF][L]
constexpr size_t OFF_OB    = OFF_VT  + (size_t)M_ * NF * 2;      // attn out bf16 [M][NF]
constexpr size_t OFF_T     = OFF_OB  + (size_t)M_ * NF * 2;      // t fp32 [M][NF]

// ---- attention LDS layout (bytes), TDM-padded rows for conflict-free ds reads ----
constexpr unsigned K_STRIDE = 264;                       // 256 + 8 pad elements
constexpr unsigned V_STRIDE = 72;                        // 64 + 8 pad elements
constexpr unsigned K_BYTES  = 64  * K_STRIDE * 2;        // 33792
constexpr unsigned V_BYTES  = 256 * V_STRIDE * 2;        // 36864
constexpr unsigned OFF_K0   = 0;
constexpr unsigned OFF_K1   = K_BYTES;
constexpr unsigned OFF_V0   = 2 * K_BYTES;
constexpr unsigned OFF_V1   = 2 * K_BYTES + V_BYTES;
constexpr unsigned OFF_PL   = 2 * K_BYTES + 2 * V_BYTES; // per-wave P tiles
constexpr unsigned SMEM_ATTN = OFF_PL + 8 * 16 * 72 * 2; // 159744 -> 2 blocks / 320KB WGP

DEVI v8f wmma_bf16(v16bf a, v16bf b, v8f c) {
  return __builtin_amdgcn_wmma_f32_16x16x32_bf16(false, a, false, b, (short)0, c,
                                                 false, false);
}

DEVI v16bf combine(v8bf lo, v8bf hi) {
  v16bf r;
#pragma unroll
  for (int i = 0; i < 8; ++i) { r[i] = lo[i]; r[i + 8] = hi[i]; }
  return r;
}

DEVI v16bf load_a_f32(const float* row, int c0, int c1) {
  v4f a0 = *(const v4f*)(row + c0);
  v4f a1 = *(const v4f*)(row + c0 + 4);
  v4f b0 = *(const v4f*)(row + c1);
  v4f b1 = *(const v4f*)(row + c1 + 4);
  v16bf r;
#pragma unroll
  for (int i = 0; i < 4; ++i) {
    r[i]      = (bf16)a0[i];
    r[4 + i]  = (bf16)a1[i];
    r[8 + i]  = (bf16)b0[i];
    r[12 + i] = (bf16)b1[i];
  }
  return r;
}

// ---- Tensor Data Mover: 2D tile load, bf16 elements, optional LDS row padding ----
// lds_off        : LDS byte address of tile start
// gaddr          : global byte address of tile start
// tile_d0/tile_d1: tile width (contiguous elements) / height (rows)
// stride_el      : tensor_dim0_stride in elements
// pad_iv/pad_amt : D# encodings (pad every 2<<pad_iv dwords by pad_amt+1 dwords)
DEVI void tdm_load_2d(unsigned lds_off, const void* gaddr,
                      unsigned tile_d0, unsigned tile_d1, unsigned stride_el,
                      unsigned pad_iv, unsigned pad_amt, bool pad_en) {
  unsigned long long ga = (unsigned long long)gaddr;
  v4u g0;
  g0[0] = 1u;                                            // count=1, user descriptor
  g0[1] = lds_off;                                       // lds_addr
  g0[2] = (unsigned)(ga & 0xFFFFFFFFu);                  // global_addr[31:0]
  g0[3] = (unsigned)((ga >> 32) & 0x01FFFFFFu) | (2u << 30);  // addr[56:32] | type=2
  v8i g1;
  unsigned w0 = (1u << 16);                              // data_size = 2 bytes
  if (pad_en) w0 |= (1u << 20) | (pad_iv << 22) | (pad_amt << 25);
  g1[0] = (int)w0;                                       // mask=0 (no cluster)
  g1[1] = (int)((tile_d0 & 0xFFFFu) << 16);              // tensor_dim0[15:0]
  g1[2] = (int)(((tile_d0 >> 16) & 0xFFFFu) | ((tile_d1 & 0xFFFFu) << 16));
  g1[3] = (int)(((tile_d1 >> 16) & 0xFFFFu) | ((tile_d0 & 0xFFFFu) << 16)); // tile_dim0
  g1[4] = (int)(tile_d1 & 0xFFFFu);                      // tile_dim1, tile_dim2=0
  g1[5] = (int)stride_el;                                // tensor_dim0_stride[31:0]
  g1[6] = 0;
  g1[7] = 0;
  v4i z4 = {0, 0, 0, 0};
#if defined(__clang_major__) && (__clang_major__ >= 23)
  v8i z8 = {0, 0, 0, 0, 0, 0, 0, 0};
  __builtin_amdgcn_tensor_load_to_lds(g0, g1, z4, z4, z8, 0);
#else
  __builtin_amdgcn_tensor_load_to_lds(g0, g1, z4, z4, 0);
#endif
}

// ---------------- prep kernels ----------------
__global__ void k_init(float* stats) {
  int i = threadIdx.x;                // 256 threads
  stats[i]       = 0.f;               // sum
  stats[256 + i] = 0.f;               // sumsq
  if (i == 0) stats[512] = 0.f;       // count
}

__global__ void k_count(const unsigned char* mask, float* cnt) {
  int i = blockIdx.x * blockDim.x + threadIdx.x;
  bool v = (i < M_) && (mask[i] != 0);
  unsigned long long bal = __ballot(v);
  if ((threadIdx.x & 31) == 0) atomicAdd(cnt, (float)__popcll(bal));
}

// transpose + convert weights: WT[e][d] = (bf16) W[d][e]
__global__ void k_wt(const float* Wq, const float* Wk, const float* Wv,
                     const float* Wt, bf16* wtbase) {
  const float* W;
  if      (blockIdx.y == 0) W = Wq;
  else if (blockIdx.y == 1) W = Wk;
  else if (blockIdx.y == 2) W = Wv;
  else                      W = Wt;
  bf16* WT = wtbase + (size_t)blockIdx.y * NF * NF;
  int idx = blockIdx.x * blockDim.x + threadIdx.x;   // 0..65535
  int d = idx >> 8, e = idx & 255;
  WT[e * NF + d] = (bf16)W[idx];
}

// ---------------- QKV projection (WMMA) ----------------
__global__ __launch_bounds__(256) void k_qkv(const float* __restrict__ x,
                                             const bf16* __restrict__ wtbase,
                                             bf16* __restrict__ qb,
                                             bf16* __restrict__ kbuf,
                                             bf16* __restrict__ vT) {
  int wave = threadIdx.x >> 5, lane = threadIdx.x & 31;
  int lm = lane & 15, lh = lane >> 4;
  int m0 = blockIdx.x * 128 + wave * 16;
  int n0 = blockIdx.y * 16;
  int z  = blockIdx.z;                                 // 0=q 1=k 2=v
  const bf16*  WT = wtbase + (size_t)z * NF * NF;
  const float* xr = x + (size_t)(m0 + lm) * NF;
  v8f acc = {};
#pragma unroll
  for (int f = 0; f < 8; ++f) {
    v16bf a = load_a_f32(xr, f * 32 + lh * 16, f * 32 + lh * 16 + 8);
    v16bf b = *(const v16bf*)(WT + (size_t)(n0 + lm) * NF + f * 32 + lh * 16);
    acc = wmma_bf16(a, b, acc);
  }
  if (z < 2) {
    bf16* out = (z == 0) ? qb : kbuf;
#pragma unroll
    for (int r = 0; r < 8; ++r)
      out[(size_t)(m0 + r + lh * 8) * NF + n0 + lm] = (bf16)acc[r];
  } else {
    int bq = m0 / L_, l0 = m0 % L_;
    v8bf pk;
#pragma unroll
    for (int r = 0; r < 8; ++r) pk[r] = (bf16)acc[r];
    *(v8bf*)(vT + ((size_t)(bq * NF + n0 + lm)) * L_ + l0 + lh * 8) = pk;
  }
}

// ---------------- flash attention: TDM double-buffered K/V + WMMA ----------------
__global__ __launch_bounds__(256) void k_attn(const bf16* __restrict__ qb,
                                              const bf16* __restrict__ kbuf,
                                              const bf16* __restrict__ vT,
                                              const unsigned char* __restrict__ mask,
                                              bf16* __restrict__ Ob) {
  extern __shared__ __align__(16) char smem[];
  int wave = threadIdx.x >> 5, lane = threadIdx.x & 31;
  int lm = lane & 15, lh = lane >> 4;
  int bq = blockIdx.x >> 5;
  int m0 = (blockIdx.x & 31) * 128 + wave * 16;      // query row within batch

  unsigned lds0 = (unsigned)(unsigned long long)(void*)smem;  // LDS byte base
  bf16* pl = (bf16*)(smem + OFF_PL) + wave * 16 * 72;         // per-wave P tile

  const v8bf* qrow = (const v8bf*)(qb + ((size_t)(bq * L_ + m0 + lm)) * NF);
  v16bf qa[8];
#pragma unroll
  for (int f = 0; f < 8; ++f) qa[f] = combine(qrow[f * 4 + lh * 2], qrow[f * 4 + lh * 2 + 1]);

  float runmax[8], runsum[8];
#pragma unroll
  for (int r = 0; r < 8; ++r) { runmax[r] = NEGV; runsum[r] = 0.f; }
  v8f zero = {};
  v8f o[16];
#pragma unroll
  for (int dt = 0; dt < 16; ++dt) o[dt] = zero;

  const unsigned char* mrow = mask + (size_t)bq * L_;
  const int NIT = L_ / 64;

  // prologue: stage key-block 0 into buffer 0 (each wave DMAs its slice)
  //   K slice: 8 rows x 256 el, LDS rows padded 256->264 el (pad 4dw / 128dw)
  //   V slice: 32 rows x 64 el, LDS rows padded 64->72 el   (pad 4dw / 32dw)
  tdm_load_2d(lds0 + OFF_K0 + (unsigned)wave * 8 * K_STRIDE * 2,
              kbuf + ((size_t)(bq * L_ + wave * 8)) * NF, 256, 8, 256, 6, 3, true);
  tdm_load_2d(lds0 + OFF_V0 + (unsigned)wave * 32 * V_STRIDE * 2,
              vT + ((size_t)(bq * NF + wave * 32)) * L_, 64, 32, L_, 4, 3, true);
  __builtin_amdgcn_s_wait_tensorcnt(0);
  __syncthreads();

  for (int it = 0; it < NIT; ++it) {
    int kb0 = it * 64;
    const bf16* kls = (const bf16*)(smem + ((it & 1) ? OFF_K1 : OFF_K0));
    const bf16* vls = (const bf16*)(smem + ((it & 1) ? OFF_V1 : OFF_V0));
    if (it + 1 < NIT) {   // async-stage next block into the other buffer
      unsigned ko = (it & 1) ? OFF_K0 : OFF_K1;
      unsigned vo = (it & 1) ? OFF_V0 : OFF_V1;
      tdm_load_2d(lds0 + ko + (unsigned)wave * 8 * K_STRIDE * 2,
                  kbuf + ((size_t)(bq * L_ + kb0 + 64 + wave * 8)) * NF,
                  256, 8, 256, 6, 3, true);
      tdm_load_2d(lds0 + vo + (unsigned)wave * 32 * V_STRIDE * 2,
                  vT + ((size_t)(bq * NF + wave * 32)) * L_ + kb0 + 64,
                  64, 32, L_, 4, 3, true);
      __builtin_prefetch(mrow + kb0 + 64, 0, 1);
    }
    // ---- S = Q K^T for 4 key tiles of 16 (K from LDS) ----
    v8f st[4];
#pragma unroll
    for (int jt = 0; jt < 4; ++jt) {
      int kl = jt * 16 + lm;
      const bf16* krow = kls + (size_t)kl * K_STRIDE + lh * 16;
      v8f s = zero;
#pragma unroll
      for (int f = 0; f < 8; ++f) {
        v16bf bb = *(const v16bf*)(krow + f * 32);
        s = wmma_bf16(qa[f], bb, s);
      }
      bool valid = (mrow[kb0 + kl] != 0);
#pragma unroll
      for (int r = 0; r < 8; ++r) s[r] = valid ? s[r] : NEGV;
      st[jt] = s;
    }
    // ---- online softmax (rows live across 16-lane halves) ----
    float tmax[8];
#pragma unroll
    for (int r = 0; r < 8; ++r)
      tmax[r] = fmaxf(fmaxf(st[0][r], st[1][r]), fmaxf(st[2][r], st[3][r]));
#pragma unroll
    for (int d = 1; d < 16; d <<= 1)
#pragma unroll
      for (int r = 0; r < 8; ++r)
        tmax[r] = fmaxf(tmax[r], __shfl_xor(tmax[r], d, 32));
    float scale[8];
#pragma unroll
    for (int r = 0; r < 8; ++r) {
      float nm = fmaxf(runmax[r], tmax[r]);
      scale[r] = __expf(runmax[r] - nm);
      runmax[r] = nm;
    }
#pragma unroll
    for (int jt = 0; jt < 4; ++jt)
#pragma unroll
      for (int r = 0; r < 8; ++r)
        st[jt][r] = __expf(st[jt][r] - runmax[r]);
    float rsum[8];
#pragma unroll
    for (int r = 0; r < 8; ++r)
      rsum[r] = (st[0][r] + st[1][r]) + (st[2][r] + st[3][r]);
#pragma unroll
    for (int d = 1; d < 16; d <<= 1)
#pragma unroll
      for (int r = 0; r < 8; ++r)
        rsum[r] += __shfl_xor(rsum[r], d, 32);
#pragma unroll
    for (int r = 0; r < 8; ++r)
      runsum[r] = runsum[r] * scale[r] + rsum[r];
#pragma unroll
    for (int dt = 0; dt < 16; ++dt)
#pragma unroll
      for (int r = 0; r < 8; ++r)
        o[dt][r] *= scale[r];
    // ---- P: D-fragment -> LDS -> A-fragment (wave-private region) ----
#pragma unroll
    for (int jt = 0; jt < 4; ++jt)
#pragma unroll
      for (int r = 0; r < 8; ++r)
        pl[(r + lh * 8) * 72 + jt * 16 + lm] = (bf16)st[jt][r];
    const v8bf* prow = (const v8bf*)(pl + lm * 72);
    v16bf pa[2];
#pragma unroll
    for (int f = 0; f < 2; ++f) pa[f] = combine(prow[f * 4 + lh * 2], prow[f * 4 + lh * 2 + 1]);
    // ---- O += P V  (V from LDS, conflict-free padded rows) ----
#pragma unroll
    for (int dt = 0; dt < 16; ++dt) {
      const bf16* vrow = vls + (size_t)(dt * 16 + lm) * V_STRIDE + lh * 16;
#pragma unroll
      for (int f = 0; f < 2; ++f) {
        v16bf bb = *(const v16bf*)(vrow + f * 32);
        o[dt] = wmma_bf16(pa[f], bb, o[dt]);
      }
    }
    // next buffer staged + everyone done with current buffer
    __builtin_amdgcn_s_wait_tensorcnt(0);
    __syncthreads();
  }
  // ---- normalize and write attn output (bf16 row-major) ----
  float rinv[8];
#pragma unroll
  for (int r = 0; r < 8; ++r) rinv[r] = 1.f / runsum[r];
#pragma unroll
  for (int dt = 0; dt < 16; ++dt)
#pragma unroll
    for (int r = 0; r < 8; ++r)
      Ob[((size_t)(bq * L_ + m0 + r + lh * 8)) * NF + dt * 16 + lm] =
          (bf16)(o[dt][r] * rinv[r]);
}

// ---------------- t = attn @ Wt (WMMA) + masked BN stats ----------------
__global__ __launch_bounds__(256) void k_tgemm(const bf16* __restrict__ Ob,
                                               const bf16* __restrict__ WtT,
                                               const unsigned char* __restrict__ mask,
                                               float* __restrict__ t,
                                               float* __restrict__ stats) {
  int wave = threadIdx.x >> 5, lane = threadIdx.x & 31;
  int lm = lane & 15, lh = lane >> 4;
  int m0 = blockIdx.x * 128 + wave * 16;
  int n0 = blockIdx.y * 16;
  const v8bf* arow = (const v8bf*)(Ob + (size_t)(m0 + lm) * NF);
  v8f acc = {};
#pragma unroll
  for (int f = 0; f < 8; ++f) {
    v16bf a = combine(arow[f * 4 + lh * 2], arow[f * 4 + lh * 2 + 1]);
    v16bf b = *(const v16bf*)(WtT + (size_t)(n0 + lm) * NF + f * 32 + lh * 16);
    acc = wmma_bf16(a, b, acc);
  }
  float s1 = 0.f, s2 = 0.f;
#pragma unroll
  for (int r = 0; r < 8; ++r) {
    int row   = m0 + r + lh * 8;
    float val = acc[r];
    t[(size_t)row * NF + n0 + lm] = val;
    float w = (mask[row] != 0) ? 1.f : 0.f;
    s1 += val * w;
    s2 += val * val * w;
  }
  s1 += __shfl_xor(s1, 16, 32);
  s2 += __shfl_xor(s2, 16, 32);
  if (lh == 0) {
    atomicAdd(&stats[n0 + lm],       s1);
    atomicAdd(&stats[256 + n0 + lm], s2);
  }
}

// ---------------- BN-ReLU + residual + mask ----------------
__global__ void k_final(const float* __restrict__ x,
                        const unsigned char* __restrict__ mask,
                        const float* __restrict__ gamma,
                        const float* __restrict__ beta,
                        const float* __restrict__ t,
                        const float* __restrict__ stats,
                        float* __restrict__ out, int n) {
  float cnt = stats[512];
  for (int i = blockIdx.x * blockDim.x + threadIdx.x; i < n;
       i += gridDim.x * blockDim.x) {
    int c   = i & 255;
    int row = i >> 8;
    float mean = stats[c] / cnt;
    float var  = stats[256 + c] / cnt - mean * mean;
    float y = gamma[c] * (t[i] - mean) * rsqrtf(var + EPS) + beta[c];
    y = fmaxf(y, 0.f);
    out[i] = (mask[row] != 0) ? (x[i] + y) : 0.f;
  }
}

extern "C" void kernel_launch(void* const* d_in, const int* in_sizes, int n_in,
                              void* d_out, int out_size, void* d_ws, size_t ws_size,
                              hipStream_t stream) {
  const float*         x     = (const float*)d_in[0];
  const unsigned char* mask  = (const unsigned char*)d_in[1];
  const float*         Wq    = (const float*)d_in[2];
  const float*         Wk    = (const float*)d_in[3];
  const float*         Wv    = (const float*)d_in[4];
  const float*         Wt    = (const float*)d_in[5];
  const float*         gamma = (const float*)d_in[6];
  const float*         beta  = (const float*)d_in[7];
  char*  ws    = (char*)d_ws;
  float* stats = (float*)(ws + OFF_STATS);
  bf16*  wtb   = (bf16*)(ws + OFF_WQT);
  bf16*  qb    = (bf16*)(ws + OFF_QB);
  bf16*  kbf   = (bf16*)(ws + OFF_KB);
  bf16*  vT    = (bf16*)(ws + OFF_VT);
  bf16*  Ob    = (bf16*)(ws + OFF_OB);
  float* t     = (float*)(ws + OFF_T);
  float* out   = (float*)d_out;

  k_init <<<1, 256, 0, stream>>>(stats);
  k_wt   <<<dim3(256, 4), 256, 0, stream>>>(Wq, Wk, Wv, Wt, wtb);
  k_count<<<M_ / 256, 256, 0, stream>>>(mask, stats + 512);
  k_qkv  <<<dim3(M_ / 128, NF / 16, 3), 256, 0, stream>>>(x, wtb, qb, kbf, vT);
  k_attn <<<dim3(B_ * (L_ / 128)), 256, SMEM_ATTN, stream>>>(qb, kbf, vT, mask, Ob);
  k_tgemm<<<dim3(M_ / 128, NF / 16), 256, 0, stream>>>(
      Ob, wtb + 3 * (size_t)NF * NF, mask, t, stats);
  k_final<<<8192, 256, 0, stream>>>(x, mask, gamma, beta, t, stats, out, M_ * NF);
}
